// PursuitGraphModel_19310172962844
// MI455X (gfx1250) — compile-verified, matrix operand order checked by reading, not run on hardware
//
#include <hip/hip_runtime.h>
#include <hip/hip_bf16.h>

#define N_NODES  100000
#define E_EDGES  1600000
#define DIN      6
#define D        64
#define LAYERS   4
#define SELF_DEG 16

typedef __attribute__((ext_vector_type(16))) __bf16 bf16x16;
typedef __attribute__((ext_vector_type(8)))  float  f32x8;

// Native CDNA5 f32 scatter-add: GLOBAL_ATOMIC_ADD_F32, no-return form
// (tracked by STOREcnt; implicit wait-idle at S_ENDPGM covers completion).
__device__ __forceinline__ void global_fadd(float* p, float v) {
    asm volatile("global_atomic_add_f32 %0, %1, off"
                 :
                 : "v"(p), "v"(v)
                 : "memory");
}

// ---------------------------------------------------------------- utilities
__global__ void zero_f32_kernel(float* p, int n) {
    int i = blockIdx.x * blockDim.x + threadIdx.x;
    if (i < n) p[i] = 0.0f;
}
__global__ void zero_i32_kernel(int* p, int n) {
    int i = blockIdx.x * blockDim.x + threadIdx.x;
    if (i < n) p[i] = 0;
}

// in-degree counts (shared by all layers)
__global__ void deg_kernel(const int* __restrict__ dst, int* __restrict__ cnt, int E) {
    int e = blockIdx.x * blockDim.x + threadIdx.x;
    if (e < E) atomicAdd(&cnt[dst[e]], 1);
}

// W[k][n] (f32, 64x64) -> WT[n][k] (bf16) so WMMA B-operand loads are K-contiguous
__global__ void wconv_kernel(const float* __restrict__ W, __bf16* __restrict__ WT) {
    int i = blockIdx.x * blockDim.x + threadIdx.x;
    if (i < D * D) {
        int k = i >> 6, n = i & 63;
        WT[n * D + k] = (__bf16)W[k * D + n];
    }
}

// ------------------------------------------------------- edge aggregation
// scalar version (layer 0, width DIN=6)
__global__ void agg_scalar_kernel(const float* __restrict__ h, const int* __restrict__ src,
                                  const int* __restrict__ dst, float* __restrict__ agg,
                                  int E, int d) {
    int i = blockIdx.x * blockDim.x + threadIdx.x;
    if (i >= E * d) return;
    int e = i / d, f = i - e * d;
    global_fadd(&agg[dst[e] * d + f], h[src[e] * d + f]);
}

// float4 version (width 64): one thread per (edge, 4-feature chunk)
__global__ void agg_vec4_kernel(const float* __restrict__ h, const int* __restrict__ src,
                                const int* __restrict__ dst, float* __restrict__ agg, int E) {
    int i = blockIdx.x * blockDim.x + threadIdx.x;
    if (i >= E * (D / 4)) return;
    int e = i >> 4;
    int c = (i & 15) * 4;
    int s = src[e], t = dst[e];
    const float4 v = *(const float4*)(h + (size_t)s * D + c);
    float* a = agg + (size_t)t * D + c;
    global_fadd(a + 0, v.x);
    global_fadd(a + 1, v.y);
    global_fadd(a + 2, v.z);
    global_fadd(a + 3, v.w);
}

// ------------------------------------------------------------- layer 0 (K=6)
__global__ void layer0_kernel(const float* __restrict__ x, const float* __restrict__ agg6,
                              const int* __restrict__ cnt,
                              const float* __restrict__ W0l, const float* __restrict__ W0r,
                              const float* __restrict__ b0, float* __restrict__ hout) {
    int i = blockIdx.x * blockDim.x + threadIdx.x;
    if (i >= N_NODES * D) return;
    int n = i >> 6, d = i & 63;
    float inv = 1.0f / fmaxf((float)cnt[n], 1.0f);
    float acc = b0[d];
#pragma unroll
    for (int k = 0; k < DIN; ++k) {
        acc += (agg6[n * DIN + k] * inv) * W0l[k * D + d];
        acc += x[n * DIN + k] * W0r[k * D + d];
    }
    hout[i] = fmaxf(acc, 0.0f);  // ReLU after layer 0
}

// ------------------------------------------- layers 1..3: WMMA 16x16x32 bf16
// One wave per 16-row tile. hout = act( mean@Wl + hin@Wr + b ),
// mean = agg * 1/max(cnt,1). N_NODES % 16 == 0.
// A layout (16-bit A 16x32, wave32): lanes 0-15 & 16-31 hold rows M=lane&15;
//   element e of the v16 maps to K = kb*32 + (e<8?0:16) + (lane>>4)*8 + (e&7).
// B layout symmetric with lane = column N. C/D: element j -> M=j+8*(lane>>4), N=lane&15.
__global__ void __launch_bounds__(256)
sage_wmma_kernel(const float* __restrict__ hin, const float* __restrict__ agg,
                 const int* __restrict__ cnt,
                 const __bf16* __restrict__ WTl,  // [n][k] bf16
                 const __bf16* __restrict__ WTr,  // [n][k] bf16
                 const float* __restrict__ bias, float* __restrict__ hout, int do_relu) {
    const int wave = (blockIdx.x * blockDim.x + threadIdx.x) >> 5;
    const int lane = threadIdx.x & 31;
    const int numTiles = N_NODES / 16;
    if (wave >= numTiles) return;          // wave-uniform: EXEC stays all-ones
    const int half = lane >> 4;
    const int r = lane & 15;
    const int row = wave * 16 + r;

    const float inv = 1.0f / fmaxf((float)cnt[row], 1.0f);
    const float* aggRow = agg + (size_t)row * D;
    const float* hRow = hin + (size_t)row * D;

    // A operands: mean and h, for both K-blocks, converted f32 -> bf16
    bf16x16 aM[2], aH[2];
#pragma unroll
    for (int kb = 0; kb < 2; ++kb) {
#pragma unroll
        for (int e = 0; e < 16; ++e) {
            int k = kb * 32 + ((e < 8) ? 0 : 16) + half * 8 + (e & 7);
            aM[kb][e] = (__bf16)(aggRow[k] * inv);
            aH[kb][e] = (__bf16)hRow[k];
        }
    }

#pragma unroll
    for (int nt = 0; nt < 4; ++nt) {
        const int col = nt * 16 + r;
        const __bf16* wl = WTl + (size_t)col * D;   // K-contiguous per lane
        const __bf16* wr = WTr + (size_t)col * D;
        f32x8 c = {0.f, 0.f, 0.f, 0.f, 0.f, 0.f, 0.f, 0.f};
#pragma unroll
        for (int kb = 0; kb < 2; ++kb) {
            bf16x16 bl, br;
#pragma unroll
            for (int e = 0; e < 16; ++e) {
                int k = kb * 32 + ((e < 8) ? 0 : 16) + half * 8 + (e & 7);
                bl[e] = wl[k];
                br[e] = wr[k];
            }
            c = __builtin_amdgcn_wmma_f32_16x16x32_bf16(false, aM[kb], false, bl,
                                                        (short)0, c, false, false);
            c = __builtin_amdgcn_wmma_f32_16x16x32_bf16(false, aH[kb], false, br,
                                                        (short)0, c, false, false);
        }
        const float bv = bias[col];
#pragma unroll
        for (int j = 0; j < 8; ++j) {
            int orow = wave * 16 + j + 8 * half;
            float v = c[j] + bv;
            if (do_relu) v = fmaxf(v, 0.0f);
            hout[(size_t)orow * D + col] = v;
        }
    }
}

// --------------------------------------------------------------- head
// out[0..15]=logits, out[16]=value, out[17..32]=neighbors (as float)
__global__ void head_kernel(const float* __restrict__ h, const int* __restrict__ src,
                            const int* __restrict__ dst, const int* __restrict__ self_p,
                            const float* __restrict__ Wlog, const float* __restrict__ blog,
                            const float* __restrict__ Wval, const float* __restrict__ bval,
                            float* __restrict__ out) {
    __shared__ int nbr[SELF_DEG];
    const int t = threadIdx.x;
    const int self = self_p[0];
    if (t == 0) {
        int c = 0;
        for (int e = 0; e < E_EDGES && c < SELF_DEG; ++e)
            if (src[e] == self) nbr[c++] = dst[e];
        for (; c < SELF_DEG; ++c) nbr[c] = dst[0];   // nonzero fill_value=0 semantics
    }
    __syncthreads();
    const float* hs = h + (size_t)self * D;
    if (t < SELF_DEG) {
        const float* hn = h + (size_t)nbr[t] * D;
        float acc = blog[0];
        for (int d = 0; d < D; ++d) acc += hs[d] * Wlog[d] + hn[d] * Wlog[D + d];
        out[t] = acc;
        out[17 + t] = (float)nbr[t];
    }
    if (t == SELF_DEG) {
        float acc = bval[0];
        for (int d = 0; d < D; ++d) acc += hs[d] * Wval[d];
        out[16] = acc;
    }
}

// ------------------------------------------------------------------ launch
extern "C" void kernel_launch(void* const* d_in, const int* in_sizes, int n_in,
                              void* d_out, int out_size, void* d_ws, size_t ws_size,
                              hipStream_t stream) {
    const float* x    = (const float*)d_in[0];
    const int*   ei   = (const int*)d_in[1];
    const int*   self = (const int*)d_in[2];
    const float* W0l  = (const float*)d_in[3];
    const float* W0r  = (const float*)d_in[4];
    const float* b0   = (const float*)d_in[5];
    const float* Wl   = (const float*)d_in[6];   // [3][64][64]
    const float* Wr   = (const float*)d_in[7];   // [3][64][64]
    const float* b    = (const float*)d_in[8];   // [3][64]
    const float* Wlog = (const float*)d_in[9];
    const float* blog = (const float*)d_in[10];
    const float* Wval = (const float*)d_in[11];
    const float* bval = (const float*)d_in[12];

    const int* src = ei;
    const int* dst = ei + E_EDGES;

    char* ws = (char*)d_ws;
    size_t o = 0;
    int*   cnt = (int*)(ws + o);  o += (((size_t)N_NODES * 4) + 255) & ~(size_t)255;
    float* agg = (float*)(ws + o); o += (size_t)N_NODES * D * 4;
    float* hA  = (float*)(ws + o); o += (size_t)N_NODES * D * 4;
    float* hB  = (float*)(ws + o); o += (size_t)N_NODES * D * 4;
    __bf16* WT = (__bf16*)(ws + o); o += (size_t)6 * D * D * 2;  // [layer][l/r][64*64]

    const int TB = 256;
    // --- degrees (layer invariant) ---
    zero_i32_kernel<<<(N_NODES + TB - 1) / TB, TB, 0, stream>>>(cnt, N_NODES);
    deg_kernel<<<(E_EDGES + TB - 1) / TB, TB, 0, stream>>>(dst, cnt, E_EDGES);

    // --- bf16 transposed weights for layers 1..3 ---
    for (int i = 0; i < 3; ++i) {
        wconv_kernel<<<(D * D + TB - 1) / TB, TB, 0, stream>>>(Wl + (size_t)i * D * D,
                                                               WT + (size_t)(2 * i) * D * D);
        wconv_kernel<<<(D * D + TB - 1) / TB, TB, 0, stream>>>(Wr + (size_t)i * D * D,
                                                               WT + (size_t)(2 * i + 1) * D * D);
    }

    // --- layer 0 (DIN=6 -> 64) ---
    zero_f32_kernel<<<(N_NODES * DIN + TB - 1) / TB, TB, 0, stream>>>(agg, N_NODES * DIN);
    agg_scalar_kernel<<<(E_EDGES * DIN + TB - 1) / TB, TB, 0, stream>>>(x, src, dst, agg,
                                                                        E_EDGES, DIN);
    layer0_kernel<<<(N_NODES * D + TB - 1) / TB, TB, 0, stream>>>(x, agg, cnt, W0l, W0r, b0, hA);

    // --- layers 1..3 (64 -> 64, WMMA) ---
    float* hin = hA;
    float* hout = hB;
    const int numTiles = N_NODES / 16;
    const int gemmBlocks = (numTiles * 32 + TB - 1) / TB;
    for (int i = 0; i < 3; ++i) {
        zero_f32_kernel<<<(N_NODES * D + TB - 1) / TB, TB, 0, stream>>>(agg, N_NODES * D);
        agg_vec4_kernel<<<(E_EDGES * (D / 4) + TB - 1) / TB, TB, 0, stream>>>(hin, src, dst,
                                                                              agg, E_EDGES);
        sage_wmma_kernel<<<gemmBlocks, TB, 0, stream>>>(hin, agg, cnt,
                                                        WT + (size_t)(2 * i) * D * D,
                                                        WT + (size_t)(2 * i + 1) * D * D,
                                                        b + (size_t)i * D, hout,
                                                        (i < 2) ? 1 : 0);
        float* tmp = hin; hin = hout; hout = tmp;
    }

    // --- head ---
    head_kernel<<<1, 32, 0, stream>>>(hin, src, dst, self, Wlog, blog, Wval, bval,
                                      (float*)d_out);
}
